// GRUSwishNorm_43791486550536
// MI455X (gfx1250) — matrix operand order repaired and checked by reading, not compiled, not run on hardware
//
#include <hip/hip_runtime.h>

// ---------------- problem constants ----------------
#define BB   8
#define TT   2048
#define DD   1024
#define BT   (BB * TT)        // 16384 rows
#define NCH  32               // scan chunks over T
#define CHLEN (TT / NCH)      // 64

// ---------------- GEMM tiling ----------------
#define BMT 128               // block tile M (8 waves: 4 waveM x 2 waveN)
#define BNT 32                // block tile N
#define KT  32                // K step (one bf16 WMMA K)
#define ASTR 40               // ushorts per A fragment row (2*16 + 8 pad, 80B: 16B-aligned, bank-spread)
#define BSTR 40               // ushorts per B fragment row

typedef __attribute__((ext_vector_type(16))) __bf16 v16bf;
typedef __attribute__((ext_vector_type(8)))  float  v8f;

// fp32 -> bf16 round-to-nearest-even, and back
__device__ __forceinline__ unsigned short f2bf(float f) {
  unsigned int u = __float_as_uint(f);
  u += 0x7FFFu + ((u >> 16) & 1u);
  return (unsigned short)(u >> 16);
}
__device__ __forceinline__ float bf2f(unsigned short h) {
  return __uint_as_float(((unsigned int)h) << 16);
}

union FragBF {
  unsigned short s[16];
  uint4          q[2];
  v16bf          v;
};

// load a full 16-elem bf16 fragment: two ds_load_b128
__device__ __forceinline__ v16bf load_frag(const unsigned short* p) {
  FragBF f;
  f.q[0] = *(const uint4*)(p);
  f.q[1] = *(const uint4*)(p + 8);
  return f.v;
}

// split 8 floats into packed bf16 hi / lo uint4's (for ds_store_b128)
__device__ __forceinline__ void split8(const float* v, uint4& hi, uint4& lo) {
  unsigned short h[8], l[8];
#pragma unroll
  for (int i = 0; i < 8; ++i) {
    h[i] = f2bf(v[i]);
    l[i] = f2bf(v[i] - bf2f(h[i]));
  }
  hi.x = (unsigned)h[0] | ((unsigned)h[1] << 16);
  hi.y = (unsigned)h[2] | ((unsigned)h[3] << 16);
  hi.z = (unsigned)h[4] | ((unsigned)h[5] << 16);
  hi.w = (unsigned)h[6] | ((unsigned)h[7] << 16);
  lo.x = (unsigned)l[0] | ((unsigned)l[1] << 16);
  lo.y = (unsigned)l[2] | ((unsigned)l[3] << 16);
  lo.z = (unsigned)l[4] | ((unsigned)l[5] << 16);
  lo.w = (unsigned)l[6] | ((unsigned)l[7] << 16);
}

// ============================================================
// Kernel 1: fused dual GEMM  zpre = x@Wz, hpre = x@Wh (+bias),
// fp32 emulated via bf16 hi/lo split (hi*hi + hi*lo + lo*hi).
// LDS tiles are stored pre-swizzled into WMMA fragment order so
// every lane fetches its fragment with 2x ds_load_b128.
// ============================================================
__global__ __launch_bounds__(256)
void gemm_dual_bf16x3(const float* __restrict__ X,
                      const float* __restrict__ Wz, const float* __restrict__ bz,
                      const float* __restrict__ Wh, const float* __restrict__ bh,
                      float* __restrict__ zpre, float* __restrict__ hpre) {
  // A fragment store: AF[m*ASTR + half*16 + e], e as in ISA 16-bit A layout
  __shared__ __align__(16) unsigned short AF_hi[BMT * ASTR];   // 10 KB
  __shared__ __align__(16) unsigned short AF_lo[BMT * ASTR];   // 10 KB
  // B fragment store: BF[n*BSTR + half*16 + e], e as in ISA 16-bit A layout
  __shared__ __align__(16) unsigned short BZ_hi[BNT * BSTR];   // 2.5 KB
  __shared__ __align__(16) unsigned short BZ_lo[BNT * BSTR];
  __shared__ __align__(16) unsigned short BH_hi[BNT * BSTR];
  __shared__ __align__(16) unsigned short BH_lo[BNT * BSTR];

  const int tid  = threadIdx.x;
  const int wave = tid >> 5;
  const int lane = tid & 31;
  const int m0 = blockIdx.x * BMT;
  const int n0 = blockIdx.y * BNT;
  const int waveM = wave >> 1;          // 0..3, each owns 32 rows
  const int waveN = wave & 1;           // 0..1, each owns 16 cols
  const int halfSel = (lane < 16) ? 0 : 1;
  const int lane15  = lane & 15;

  v8f accZ0 = {}, accZ1 = {};
  v8f accH0 = {}, accH1 = {};

  for (int k0 = 0; k0 < DD; k0 += KT) {
    // ---- stage A tile (128x32 fp32 -> bf16 hi/lo fragments) ----
    // 4096 elems; each thread handles 2 runs of 8 consecutive k.
#pragma unroll
    for (int j = 0; j < 2; ++j) {
      const int idx = tid * 8 + j * 2048;       // 0..4095
      const int row = idx >> 5;                  // 0..127
      const int col = idx & (KT - 1);            // 0,8,16,24
      const float* src = X + (size_t)(m0 + row) * DD + k0 + col;
      float v[8];
      float4 f0 = ((const float4*)src)[0];
      float4 f1 = ((const float4*)src)[1];
      v[0]=f0.x; v[1]=f0.y; v[2]=f0.z; v[3]=f0.w;
      v[4]=f1.x; v[5]=f1.y; v[6]=f1.z; v[7]=f1.w;
      uint4 hi, lo;
      split8(v, hi, lo);
      // run of 8 k's maps to 8 consecutive fragment elements:
      const int half  = (col & 8) ? 1 : 0;
      const int ebase = (col & 16) ? 8 : 0;
      const int a = row * ASTR + half * 16 + ebase;   // 16B-aligned
      *(uint4*)&AF_hi[a] = hi;
      *(uint4*)&AF_lo[a] = lo;
    }
    // ---- stage Wz/Wh tiles (32x32 each) into B fragment order ----
    {
      const int base = tid * 4;                  // 1024 elems / 256 thr
      const int kr = base >> 5;                  // 0..31
      const int nc = base & (BNT - 1);
      const int half = (kr >> 3) & 1;
      const int e    = (kr & 7) | ((kr >> 4) << 3);
      const float* sz = Wz + (size_t)(k0 + kr) * DD + n0 + nc;
      const float* sh = Wh + (size_t)(k0 + kr) * DD + n0 + nc;
#pragma unroll
      for (int i = 0; i < 4; ++i) {
        const int a = (nc + i) * BSTR + half * 16 + e;
        float vz = sz[i];
        unsigned short hz = f2bf(vz);
        BZ_hi[a] = hz;
        BZ_lo[a] = f2bf(vz - bf2f(hz));
        float vh = sh[i];
        unsigned short hh = f2bf(vh);
        BH_hi[a] = hh;
        BH_lo[a] = f2bf(vh - bf2f(hh));
      }
    }
    // prefetch next A tile (global_prefetch_b8)
    if (k0 + KT < DD) {
      __builtin_prefetch(X + (size_t)(m0 + (tid >> 1)) * DD + k0 + KT, 0, 1);
    }
    __syncthreads();

    // ---- fragment loads: 2x ds_load_b128 each ----
    const int mA0 = (waveM * 32 + lane15) * ASTR + halfSel * 16;
    const int mA1 = (waveM * 32 + 16 + lane15) * ASTR + halfSel * 16;
    const int nB  = (waveN * 16 + lane15) * BSTR + halfSel * 16;

    v16bf a0h = load_frag(&AF_hi[mA0]);
    v16bf a0l = load_frag(&AF_lo[mA0]);
    v16bf a1h = load_frag(&AF_hi[mA1]);
    v16bf a1l = load_frag(&AF_lo[mA1]);
    v16bf bzh = load_frag(&BZ_hi[nB]);
    v16bf bzl = load_frag(&BZ_lo[nB]);
    v16bf bhh = load_frag(&BH_hi[nB]);
    v16bf bhl = load_frag(&BH_lo[nB]);

    // 3-term bf16 split per output per M-subtile: 12 WMMAs / K-step
    accZ0 = __builtin_amdgcn_wmma_f32_16x16x32_bf16(false, a0h, false, bzh, (short)0, accZ0, false, false);
    accZ0 = __builtin_amdgcn_wmma_f32_16x16x32_bf16(false, a0h, false, bzl, (short)0, accZ0, false, false);
    accZ0 = __builtin_amdgcn_wmma_f32_16x16x32_bf16(false, a0l, false, bzh, (short)0, accZ0, false, false);
    accZ1 = __builtin_amdgcn_wmma_f32_16x16x32_bf16(false, a1h, false, bzh, (short)0, accZ1, false, false);
    accZ1 = __builtin_amdgcn_wmma_f32_16x16x32_bf16(false, a1h, false, bzl, (short)0, accZ1, false, false);
    accZ1 = __builtin_amdgcn_wmma_f32_16x16x32_bf16(false, a1l, false, bzh, (short)0, accZ1, false, false);
    accH0 = __builtin_amdgcn_wmma_f32_16x16x32_bf16(false, a0h, false, bhh, (short)0, accH0, false, false);
    accH0 = __builtin_amdgcn_wmma_f32_16x16x32_bf16(false, a0h, false, bhl, (short)0, accH0, false, false);
    accH0 = __builtin_amdgcn_wmma_f32_16x16x32_bf16(false, a0l, false, bhh, (short)0, accH0, false, false);
    accH1 = __builtin_amdgcn_wmma_f32_16x16x32_bf16(false, a1h, false, bhh, (short)0, accH1, false, false);
    accH1 = __builtin_amdgcn_wmma_f32_16x16x32_bf16(false, a1h, false, bhl, (short)0, accH1, false, false);
    accH1 = __builtin_amdgcn_wmma_f32_16x16x32_bf16(false, a1l, false, bhh, (short)0, accH1, false, false);
    __syncthreads();
  }

  // ---- epilogue: C layout (elem v: M = v + (lane<16?0:8), N = lane&15) ----
  const int n    = n0 + waveN * 16 + lane15;
  const int mOff = halfSel * 8;
  const float bzv = bz[n];
  const float bhv = bh[n];
#pragma unroll
  for (int v = 0; v < 8; ++v) {
    const size_t r0 = (size_t)(m0 + waveM * 32 + mOff + v) * DD + n;
    const size_t r1 = (size_t)(m0 + waveM * 32 + 16 + mOff + v) * DD + n;
    zpre[r0] = accZ0[v] + bzv;
    zpre[r1] = accZ1[v] + bzv;
    hpre[r0] = accH0[v] + bhv;
    hpre[r1] = accH1[v] + bhv;
  }
}

// ============================================================
// Kernel 2a: per-chunk composition (Aprod, Bcomp) of h_t = a h + b
// ============================================================
__global__ __launch_bounds__(256)
void scan_passA(const float* __restrict__ zpre, const float* __restrict__ hpre,
                float* __restrict__ Aprod, float* __restrict__ Bcomp) {
  const int id    = blockIdx.x * 256 + threadIdx.x;   // chains*chunks
  const int chain = id & (BB * DD - 1);               // 8192 chains
  const int c     = id >> 13;
  const int b     = chain >> 10;
  const int d     = chain & (DD - 1);
  size_t base = ((size_t)(b * TT + c * CHLEN)) * DD + d;
  float A = 1.f, Bc = 0.f;
  for (int t = 0; t < CHLEN; ++t) {
    float zp = zpre[base + (size_t)t * DD];
    float hp = hpre[base + (size_t)t * DD];
    float z  = 1.f / (1.f + __expf(-zp));
    float a  = 1.f - z;
    A  = A * a;
    Bc = a * Bc + z * hp;
  }
  Aprod[(size_t)c * (BB * DD) + chain] = A;
  Bcomp[(size_t)c * (BB * DD) + chain] = Bc;
}

// ============================================================
// Kernel 2b: sequential prefix over 32 chunk summaries per chain.
// Rewrites Bcomp[c] with the carry-in h value for chunk c.
// ============================================================
__global__ __launch_bounds__(256)
void scan_combine(const float* __restrict__ Aprod, float* __restrict__ Bcomp) {
  const int chain = blockIdx.x * 256 + threadIdx.x;   // 8192
  float carry = 0.f;
  for (int c = 0; c < NCH; ++c) {
    const size_t idx = (size_t)c * (BB * DD) + chain;
    float A  = Aprod[idx];
    float Bc = Bcomp[idx];
    Bcomp[idx] = carry;               // carry-in for this chunk
    carry = A * carry + Bc;
  }
}

// ============================================================
// Kernel 2c: replay each chunk from its carry-in, write h to d_out
// ============================================================
__global__ __launch_bounds__(256)
void scan_passC(const float* __restrict__ zpre, const float* __restrict__ hpre,
                const float* __restrict__ Carry, float* __restrict__ hout) {
  const int id    = blockIdx.x * 256 + threadIdx.x;
  const int chain = id & (BB * DD - 1);
  const int c     = id >> 13;
  const int b     = chain >> 10;
  const int d     = chain & (DD - 1);
  size_t base = ((size_t)(b * TT + c * CHLEN)) * DD + d;
  float h = Carry[(size_t)c * (BB * DD) + chain];
  for (int t = 0; t < CHLEN; ++t) {
    float zp = zpre[base + (size_t)t * DD];
    float hp = hpre[base + (size_t)t * DD];
    float z  = 1.f / (1.f + __expf(-zp));
    h = (1.f - z) * h + z * hp;
    hout[base + (size_t)t * DD] = h;
  }
}

// ============================================================
// Kernel 3: in-place Swish(beta) + LayerNorm over D per row
// ============================================================
__global__ __launch_bounds__(256)
void swish_ln(float* __restrict__ out, const float* __restrict__ sb,
              const float* __restrict__ gamma, const float* __restrict__ betap) {
  __shared__ float rsum[256];
  __shared__ float rsq[256];
  const int row = blockIdx.x;
  const int tid = threadIdx.x;
  const float beta = sb[0];

  float y[4];
  float s1 = 0.f, s2 = 0.f;
#pragma unroll
  for (int i = 0; i < 4; ++i) {
    const int d = tid + i * 256;
    float h = out[(size_t)row * DD + d];
    float s = beta * h;
    float yy = s / (1.f + __expf(-s));   // s * sigmoid(s)
    y[i] = yy;
    s1 += yy;
    s2 += yy * yy;
  }
  rsum[tid] = s1;
  rsq[tid]  = s2;
  __syncthreads();
  for (int st = 128; st > 0; st >>= 1) {
    if (tid < st) {
      rsum[tid] += rsum[tid + st];
      rsq[tid]  += rsq[tid + st];
    }
    __syncthreads();
  }
  const float mu  = rsum[0] * (1.f / DD);
  const float var = rsq[0] * (1.f / DD) - mu * mu;
  const float inv = rsqrtf(var + 1e-5f);
#pragma unroll
  for (int i = 0; i < 4; ++i) {
    const int d = tid + i * 256;
    out[(size_t)row * DD + d] = (y[i] - mu) * inv * gamma[d] + betap[d];
  }
}

// ============================================================
extern "C" void kernel_launch(void* const* d_in, const int* in_sizes, int n_in,
                              void* d_out, int out_size, void* d_ws, size_t ws_size,
                              hipStream_t stream) {
  const float* x     = (const float*)d_in[0];
  const float* Wz    = (const float*)d_in[1];
  const float* bz    = (const float*)d_in[2];
  const float* Wh    = (const float*)d_in[3];
  const float* bh    = (const float*)d_in[4];
  const float* sb    = (const float*)d_in[5];
  const float* gamma = (const float*)d_in[6];
  const float* betap = (const float*)d_in[7];
  float* out = (float*)d_out;

  float* zpre  = (float*)d_ws;                       // BT*DD f32
  float* hpre  = zpre + (size_t)BT * DD;             // BT*DD f32
  float* Aprod = hpre + (size_t)BT * DD;             // NCH*BB*DD f32
  float* Bcomp = Aprod + (size_t)NCH * BB * DD;      // NCH*BB*DD f32

  dim3 gGrid(BT / BMT, DD / BNT);
  gemm_dual_bf16x3<<<gGrid, 256, 0, stream>>>(x, Wz, bz, Wh, bh, zpre, hpre);
  scan_passA<<<(BB * DD * NCH) / 256, 256, 0, stream>>>(zpre, hpre, Aprod, Bcomp);
  scan_combine<<<(BB * DD) / 256, 256, 0, stream>>>(Aprod, Bcomp);
  scan_passC<<<(BB * DD * NCH) / 256, 256, 0, stream>>>(zpre, hpre, Bcomp, out);
  swish_ln<<<BT, 256, 0, stream>>>(out, sb, gamma, betap);
}
